// GAT_70970039599320
// MI455X (gfx1250) — matrix operand order repaired and checked by reading, not compiled, not run on hardware
//
#include <hip/hip_runtime.h>

#define NN      4096
#define NFEAT   512
#define NHID    64
#define NCLASS  32
#define NHEADS  8

typedef __attribute__((ext_vector_type(16))) __bf16         v16bf;
typedef __attribute__((ext_vector_type(8)))  float          v8f;
typedef __attribute__((ext_vector_type(8)))  unsigned short us8;

union ABu { v16bf v; us8 h[2]; unsigned short u[16]; };

__device__ __forceinline__ unsigned short f2bf(float f) {
  unsigned int u = __float_as_uint(f);
  u += 0x7fffu + ((u >> 16) & 1u);          // round-to-nearest-even
  return (unsigned short)(u >> 16);
}
__device__ __forceinline__ float lrelu(float v) { return fmaxf(v, 0.01f * v); }

__device__ __forceinline__ v8f wmma_bf16(const ABu& a, const ABu& b, v8f c) {
  return __builtin_amdgcn_wmma_f32_16x16x32_bf16(
      /*neg_a=*/false, a.v, /*neg_b=*/false, b.v,
      /*c_mod=*/(short)0, c, /*reuse_a=*/false, /*reuse_b=*/false);
}

// ---------------- prep: bf16 conversions / transposes / folded vectors ------
__global__ void k_xbf(const float* __restrict__ x, unsigned short* __restrict__ xbf) {
  int idx = blockIdx.x * 256 + threadIdx.x;
  xbf[idx] = f2bf(x[idx]);
}

// wt[h][o][f] = bf16(W[h][f][o])   (so B-operand reads are 32B contiguous in f)
__global__ void k_wt(const float* __restrict__ W, unsigned short* __restrict__ wt) {
  int idx = blockIdx.x * 256 + threadIdx.x;
  int f = idx & (NFEAT - 1); int o = (idx >> 9) & (NHID - 1); int h = idx >> 15;
  wt[idx] = f2bf(W[((size_t)h * NFEAT + f) * NHID + o]);
}

// wa1[h][f] = sum_o W[h][f][o]*a1[h][o]  (folds s1 = x @ wa1, avoiding h reads)
__global__ void k_wa(const float* __restrict__ W, const float* __restrict__ a1,
                     const float* __restrict__ a2, float* __restrict__ wa1,
                     float* __restrict__ wa2) {
  int idx = blockIdx.x * 256 + threadIdx.x;
  int f = idx & (NFEAT - 1); int h = idx >> 9;
  const float* wrow = W + ((size_t)h * NFEAT + f) * NHID;
  float u = 0.f, v = 0.f;
  #pragma unroll 8
  for (int o = 0; o < NHID; ++o) {
    float w = wrow[o];
    u += w * a1[h * NHID + o];
    v += w * a2[h * NHID + o];
  }
  wa1[idx] = u; wa2[idx] = v;
}

// w2t[c][f] = bf16(W2[f][c])
__global__ void k_w2t(const float* __restrict__ W2, unsigned short* __restrict__ w2t) {
  int idx = blockIdx.x * 256 + threadIdx.x;
  int f = idx & (NFEAT - 1); int c = idx >> 9;
  w2t[idx] = f2bf(W2[(size_t)f * NCLASS + c]);
}

// ---------------- s1[h][n], s2[h][n] = x[n,:] . wa{1,2}[h,:] (fp32) ----------
__global__ __launch_bounds__(256) void k_s12(const float* __restrict__ x,
                                             const float* __restrict__ wa1,
                                             const float* __restrict__ wa2,
                                             float* __restrict__ s1,
                                             float* __restrict__ s2) {
  int n = blockIdx.x * 256 + threadIdx.x;
  const float4* xr = (const float4*)(x + (size_t)n * NFEAT);
  const float4* w1 = (const float4*)wa1;
  const float4* w2 = (const float4*)wa2;
  float a1acc[NHEADS] = {}, a2acc[NHEADS] = {};
  for (int f4 = 0; f4 < NFEAT / 4; ++f4) {
    float4 xv = xr[f4];
    #pragma unroll
    for (int h = 0; h < NHEADS; ++h) {
      float4 wv = w1[h * (NFEAT / 4) + f4];    // uniform across lanes -> scalar loads
      a1acc[h] += xv.x * wv.x + xv.y * wv.y + xv.z * wv.z + xv.w * wv.w;
      float4 uv = w2[h * (NFEAT / 4) + f4];
      a2acc[h] += xv.x * uv.x + xv.y * uv.y + xv.z * uv.z + xv.w * uv.w;
    }
  }
  #pragma unroll
  for (int h = 0; h < NHEADS; ++h) {
    s1[(size_t)h * NN + n] = a1acc[h];
    s2[(size_t)h * NN + n] = a2acc[h];
  }
}

// ---------------- Stage A: h_bf[h][n][o] = bf16( x @ W[h] )  (WMMA) ---------
__global__ __launch_bounds__(256) void k_h(const unsigned short* __restrict__ xbf,
                                           const unsigned short* __restrict__ wt,
                                           unsigned short* __restrict__ hbf) {
  int hd = blockIdx.y;
  int wave = threadIdx.x >> 5, lane = threadIdx.x & 31;
  int g = lane >> 4, m = lane & 15;
  int i0 = blockIdx.x * 128 + wave * 16;
  const unsigned short* xrow = xbf + (size_t)(i0 + m) * NFEAT;
  const unsigned short* wth  = wt + (size_t)hd * NHID * NFEAT;
  v8f acc[4] = {};
  for (int fb = 0; fb < NFEAT; fb += 32) {
    ABu a;                                       // A: 16x32 bf16, K = fb + 16g + e
    a.h[0] = *(const us8*)(xrow + fb + 16 * g);
    a.h[1] = *(const us8*)(xrow + fb + 16 * g + 8);
    #pragma unroll
    for (int nt = 0; nt < 4; ++nt) {             // B: K = fb + 16g + e  -> contiguous row of wt
      ABu b;
      const unsigned short* br = wth + (size_t)(nt * 16 + m) * NFEAT + fb + 16 * g;
      b.h[0] = *(const us8*)br;
      b.h[1] = *(const us8*)(br + 8);
      acc[nt] = wmma_bf16(a, b, acc[nt]);
    }
  }
  #pragma unroll
  for (int nt = 0; nt < 4; ++nt)
    #pragma unroll
    for (int r = 0; r < 8; ++r) {                // D: M = r + 8g, N = m
      int i = i0 + r + 8 * g, o = nt * 16 + m;
      hbf[((size_t)hd * NN + i) * NHID + o] = f2bf(acc[nt][r]);
    }
}

// --------- Stage B (dominant): xc = lr( e @ h ), e generated on the fly -----
__global__ __launch_bounds__(256) void k_attn(const unsigned short* __restrict__ hbf,
                                              const float* __restrict__ s1,
                                              const float* __restrict__ s2,
                                              unsigned short* __restrict__ xcbf) {
  __shared__ __align__(16) unsigned short ldsB[NHID * 40]; // h tile transposed [o][j'], padded
  __shared__ float ldsS1[32];
  int hd = blockIdx.y;
  int tid = threadIdx.x;
  int wave = tid >> 5, lane = tid & 31, g = lane >> 4, m = lane & 15;
  int i0 = blockIdx.x * 128 + wave * 16;
  float s2i = s2[(size_t)hd * NN + i0 + m];
  const unsigned short* hh = hbf + (size_t)hd * NN * NHID;
  v8f acc[4] = {};
  int jj = tid >> 3, o8 = (tid & 7) * 8;
  for (int jb = 0; jb < NN; jb += 32) {
    __syncthreads();
    // cooperative coalesced load of h[jb..jb+32][0..64] -> LDS transposed
    us8 hv = *(const us8*)(hh + (size_t)(jb + jj) * NHID + o8);
    #pragma unroll
    for (int k = 0; k < 8; ++k) ldsB[(o8 + k) * 40 + jj] = hv[k];
    if (tid < 32) ldsS1[tid] = s1[(size_t)hd * NN + jb + tid];
    __syncthreads();
    // A operand: e[i, jb+j'] = lr(s1[j'] + s2[i]), contiguous K: j' = 16g + e
    ABu a;
    #pragma unroll
    for (int t = 0; t < 8; ++t) {
      int j0 = 16 * g + 2 * t;
      a.u[2 * t]     = f2bf(lrelu(ldsS1[j0]     + s2i));
      a.u[2 * t + 1] = f2bf(lrelu(ldsS1[j0 + 1] + s2i));
    }
    #pragma unroll
    for (int nt = 0; nt < 4; ++nt) {
      ABu b;                                      // B: ldsB[o][16g..16g+15] -> 2x ds_load_b128
      const unsigned short* p = &ldsB[(nt * 16 + m) * 40 + 16 * g];
      b.h[0] = *(const us8*)p;
      b.h[1] = *(const us8*)(p + 8);
      acc[nt] = wmma_bf16(a, b, acc[nt]);
    }
  }
  #pragma unroll
  for (int nt = 0; nt < 4; ++nt)
    #pragma unroll
    for (int r = 0; r < 8; ++r) {
      int i = i0 + r + 8 * g, o = nt * 16 + m;
      xcbf[(size_t)i * (NHEADS * NHID) + hd * NHID + o] = f2bf(lrelu(acc[nt][r]));
    }
}

// ---------------- Stage C: h2 = xc @ W2  (WMMA, fp32 out) -------------------
__global__ __launch_bounds__(256) void k_h2(const unsigned short* __restrict__ xcbf,
                                            const unsigned short* __restrict__ w2t,
                                            float* __restrict__ h2) {
  int wave = threadIdx.x >> 5, lane = threadIdx.x & 31, g = lane >> 4, m = lane & 15;
  int i0 = blockIdx.x * 128 + wave * 16;
  const unsigned short* xrow = xcbf + (size_t)(i0 + m) * (NHEADS * NHID);
  v8f acc[2] = {};
  for (int fb = 0; fb < NHEADS * NHID; fb += 32) {
    ABu a;
    a.h[0] = *(const us8*)(xrow + fb + 16 * g);
    a.h[1] = *(const us8*)(xrow + fb + 16 * g + 8);
    #pragma unroll
    for (int nt = 0; nt < 2; ++nt) {
      ABu b;
      const unsigned short* br = w2t + (size_t)(nt * 16 + m) * (NHEADS * NHID) + fb + 16 * g;
      b.h[0] = *(const us8*)br;
      b.h[1] = *(const us8*)(br + 8);
      acc[nt] = wmma_bf16(a, b, acc[nt]);
    }
  }
  #pragma unroll
  for (int nt = 0; nt < 2; ++nt)
    #pragma unroll
    for (int r = 0; r < 8; ++r)
      h2[(size_t)(i0 + r + 8 * g) * NCLASS + nt * 16 + m] = acc[nt][r];
}

// ---------------- final stages (algebraically collapsed) --------------------
__global__ void k_t12(const float* __restrict__ h2, const float* __restrict__ a1f,
                      const float* __restrict__ a2f, float* __restrict__ t1,
                      float* __restrict__ t2) {
  int n = blockIdx.x * 256 + threadIdx.x;
  const float* hr = h2 + (size_t)n * NCLASS;
  float u = 0.f, v = 0.f;
  #pragma unroll
  for (int c = 0; c < NCLASS; ++c) { u += hr[c] * a1f[c]; v += hr[c] * a2f[c]; }
  t1[n] = u; t2[n] = v;
}

// gpart[chunk][j] = sum_{i in chunk} lr(t1[j] + t2[i])
__global__ __launch_bounds__(256) void k_g(const float* __restrict__ t1,
                                           const float* __restrict__ t2,
                                           float* __restrict__ gpart) {
  int j = blockIdx.x * 256 + threadIdx.x;
  int ib = blockIdx.y * 512;
  float tj = t1[j];
  float sum = 0.f;
  #pragma unroll 8
  for (int i = 0; i < 512; ++i) sum += lrelu(tj + t2[ib + i]);  // t2 loads uniform -> scalar
  gpart[(size_t)blockIdx.y * NN + j] = sum;
}

__global__ void k_gsum(const float* __restrict__ gpart, float* __restrict__ gg) {
  int j = blockIdx.x * 256 + threadIdx.x;
  float s = 0.f;
  #pragma unroll
  for (int k = 0; k < 8; ++k) s += gpart[(size_t)k * NN + j];
  gg[j] = s;
}

// y_mean[c] = (1/N) sum_j g[j] h2[j,c];  out = y_mean @ Wl + bl
__global__ __launch_bounds__(256) void k_final(const float* __restrict__ gg,
                                               const float* __restrict__ h2,
                                               const float* __restrict__ Wl,
                                               const float* __restrict__ bl,
                                               float* __restrict__ out) {
  __shared__ float part[256];
  __shared__ float ym[NCLASS];
  int tid = threadIdx.x;
  int c = tid & 31, sl = tid >> 5;
  float s = 0.f;
  for (int j = sl * 512; j < sl * 512 + 512; ++j) s += gg[j] * h2[(size_t)j * NCLASS + c];
  part[tid] = s;
  __syncthreads();
  if (tid < NCLASS) {
    float y = 0.f;
    #pragma unroll
    for (int k = 0; k < 8; ++k) y += part[k * 32 + tid];
    ym[tid] = y * (1.0f / NN);
  }
  __syncthreads();
  if (tid < NCLASS) {
    float o = bl[tid];
    #pragma unroll
    for (int cc = 0; cc < NCLASS; ++cc) o += ym[cc] * Wl[cc * NCLASS + tid];
    out[tid] = o;
  }
}

extern "C" void kernel_launch(void* const* d_in, const int* in_sizes, int n_in,
                              void* d_out, int out_size, void* d_ws, size_t ws_size,
                              hipStream_t stream) {
  (void)in_sizes; (void)n_in; (void)out_size; (void)ws_size;
  const float* x   = (const float*)d_in[0];
  const float* W   = (const float*)d_in[1];
  const float* a1  = (const float*)d_in[2];
  const float* a2  = (const float*)d_in[3];
  const float* W2  = (const float*)d_in[4];
  const float* a1f = (const float*)d_in[5];
  const float* a2f = (const float*)d_in[6];
  const float* Wl  = (const float*)d_in[7];
  const float* bl  = (const float*)d_in[8];
  float* out = (float*)d_out;

  char* ws = (char*)d_ws;
  size_t off = 0;
  auto alloc = [&](size_t bytes) -> void* {
    void* p = ws + off;
    off += (bytes + 255) & ~(size_t)255;
    return p;
  };
  unsigned short* xbf   = (unsigned short*)alloc((size_t)NN * NFEAT * 2);
  unsigned short* wt    = (unsigned short*)alloc((size_t)NHEADS * NHID * NFEAT * 2);
  float*          wa1   = (float*)alloc((size_t)NHEADS * NFEAT * 4);
  float*          wa2   = (float*)alloc((size_t)NHEADS * NFEAT * 4);
  float*          s1    = (float*)alloc((size_t)NHEADS * NN * 4);
  float*          s2    = (float*)alloc((size_t)NHEADS * NN * 4);
  unsigned short* hbf   = (unsigned short*)alloc((size_t)NHEADS * NN * NHID * 2);
  unsigned short* xcbf  = (unsigned short*)alloc((size_t)NN * NHEADS * NHID * 2);
  unsigned short* w2t   = (unsigned short*)alloc((size_t)NCLASS * NHEADS * NHID * 2);
  float*          h2    = (float*)alloc((size_t)NN * NCLASS * 4);
  float*          t1    = (float*)alloc((size_t)NN * 4);
  float*          t2    = (float*)alloc((size_t)NN * 4);
  float*          gpart = (float*)alloc((size_t)8 * NN * 4);
  float*          gg    = (float*)alloc((size_t)NN * 4);

  k_xbf <<<(NN * NFEAT) / 256, 256, 0, stream>>>(x, xbf);
  k_wt  <<<(NHEADS * NHID * NFEAT) / 256, 256, 0, stream>>>(W, wt);
  k_wa  <<<(NHEADS * NFEAT) / 256, 256, 0, stream>>>(W, a1, a2, wa1, wa2);
  k_w2t <<<(NCLASS * NHEADS * NHID) / 256, 256, 0, stream>>>(W2, w2t);
  k_s12 <<<NN / 256, 256, 0, stream>>>(x, wa1, wa2, s1, s2);
  k_h   <<<dim3(NN / 128, NHEADS), 256, 0, stream>>>(xbf, wt, hbf);
  k_attn<<<dim3(NN / 128, NHEADS), 256, 0, stream>>>(hbf, s1, s2, xcbf);
  k_h2  <<<NN / 128, 256, 0, stream>>>(xcbf, w2t, h2);
  k_t12 <<<NN / 256, 256, 0, stream>>>(h2, a1f, a2f, t1, t2);
  k_g   <<<dim3(NN / 256, 8), 256, 0, stream>>>(t1, t2, gpart);
  k_gsum<<<NN / 256, 256, 0, stream>>>(gpart, gg);
  k_final<<<1, 256, 0, stream>>>(gg, h2, Wl, bl, out);
}
